// Attention_22428319220353
// MI455X (gfx1250) — matrix operand order repaired and checked by reading, not compiled, not run on hardware
//
#include <hip/hip_runtime.h>

typedef __attribute__((ext_vector_type(16))) _Float16 v16h;
typedef __attribute__((ext_vector_type(8)))  float    v8f;
typedef __attribute__((ext_vector_type(4)))  unsigned int v4u;
typedef __attribute__((ext_vector_type(8)))  int v8i;
typedef __attribute__((ext_vector_type(4)))  int v4i;

#define BH 64
#define NQ 1024
#define NK 1024
#define ND 64
#define QK (NQ * NK)
#define QD (NQ * ND)
#define KD (NK * ND)

static __device__ __forceinline__ int lane_() { return threadIdx.x & 31; }

// A fragment 16(M)x32(k) f16 from f32 source: A[m][k] = src[m*ld + k]
// wave32 layout: lane l -> row m = l%16; k offsets kb..kb+15, kb = (l<16)?0:16
static __device__ __forceinline__ v16h loadA(const float* __restrict__ src, int ld) {
  const int l = lane_();
  const int m = l & 15;
  const int kb = (l < 16) ? 0 : 16;
  const float* row = src + m * ld;
  v16h a;
#pragma unroll
  for (int j = 0; j < 8; ++j) a[j] = (_Float16)row[kb + j];
#pragma unroll
  for (int j = 0; j < 8; ++j) a[8 + j] = (_Float16)row[kb + 8 + j];
  return a;
}

// B fragment 32(k)x16(N): B[k][n] = src[n*ld + k]  (k contiguous in memory)
// lane l -> col n = l%16; k = ko + j, ko = (l<16)?0:16
static __device__ __forceinline__ v16h loadB_kcontig(const float* __restrict__ src, int ld) {
  const int l = lane_();
  const int n = l & 15;
  const int ko = (l < 16) ? 0 : 16;
  const float* row = src + n * ld + ko;
  v16h b;
#pragma unroll
  for (int j = 0; j < 16; ++j) b[j] = (_Float16)row[j];
  return b;
}

// B fragment from an LDS-resident 32(k) x 64(n) f32 tile: B[k][n] = tile[k*64 + d0 + n]
// Strided reads land in LDS (conflict-free: 16 lanes hit 16 consecutive DWORDs per row).
static __device__ __forceinline__ v16h loadB_tile(const float* tile, int d0) {
  const int l = lane_();
  const int n = l & 15;
  const int ko = (l < 16) ? 0 : 16;
  v16h b;
#pragma unroll
  for (int j = 0; j < 16; ++j) b[j] = (_Float16)tile[(ko + j) * 64 + d0 + n];
  return b;
}

// D tile 16x16 f32: lane l, reg r -> (m = mb + r, n = l%16), mb = (l<16)?0:8
static __device__ __forceinline__ void storeD(float* __restrict__ dst, int ld, v8f d) {
  const int l = lane_();
  const int n = l & 15;
  const int mb = (l < 16) ? 0 : 8;
#pragma unroll
  for (int r = 0; r < 8; ++r) dst[(mb + r) * ld + n] = d[r];
}

static __device__ __forceinline__ void storeD_add(float* __restrict__ dst, int ld, v8f d) {
  const int l = lane_();
  const int n = l & 15;
  const int mb = (l < 16) ? 0 : 8;
#pragma unroll
  for (int r = 0; r < 8; ++r) dst[(mb + r) * ld + n] += d[r];
}

#define WMMA(a, b, c) \
  __builtin_amdgcn_wmma_f32_16x16x32_f16(false, (a), false, (b), (short)0, (c), false, false)

// ---------------------------------------------------------------------------
// TDM: DMA one 32-row x 64-col f32 tile from global memory into LDS.
// Descriptor per CDNA5 ISA (D# group0/group1 bit layout):
//   group0: [1:0]=count=1, [63:32]=lds byte addr, [120:64]=global byte addr,
//           [127:126]=type=2 ("image")
//   group1: [17:16]=data_size(4B)=2, [79:48]=tensor_dim0=64,
//           [111:80]=tensor_dim1=rows remaining (OOB rows read as zero),
//           [127:112]=tile_dim0=64, [143:128]=tile_dim1=32,
//           [207:160]=tensor_dim0_stride=64
// This toolchain exposes the 6-arg builtin: (v4u, v8i, v4i, v4i, v8i, i32 cpol).
static __device__ __forceinline__ void tdm_load_32x64_f32(unsigned lds_off,
                                                          const float* gptr,
                                                          int rows_rem) {
  const unsigned long long ga = (unsigned long long)(size_t)gptr;
  v4u g0;
  g0[0] = 1u;                                             // count=1 user D#
  g0[1] = lds_off;                                        // lds_addr
  g0[2] = (unsigned)(ga & 0xffffffffu);                   // global_addr[31:0]
  g0[3] = (unsigned)((ga >> 32) & 0x1ffffffu) | (2u << 30); // addr[56:32] | type=2
  v8i g1;
  g1[0] = 2 << 16;                                        // data_size = 4 bytes
  g1[1] = 64 << 16;                                       // tensor_dim0 = 64
  g1[2] = (rows_rem & 0xffff) << 16;                      // tensor_dim1 lo
  g1[3] = ((rows_rem >> 16) & 0xffff) | (64 << 16);       // tensor_dim1 hi | tile_dim0=64
  g1[4] = 32;                                             // tile_dim1 = 32 rows
  g1[5] = 64;                                             // tensor_dim0_stride = 64
  g1[6] = 0;
  g1[7] = 0;
  v4i gz4;
  gz4[0] = 0; gz4[1] = 0; gz4[2] = 0; gz4[3] = 0;         // <=2D: groups 2/3 unused
  v8i gz8;
  gz8[0] = 0; gz8[1] = 0; gz8[2] = 0; gz8[3] = 0;
  gz8[4] = 0; gz8[5] = 0; gz8[6] = 0; gz8[7] = 0;
  __builtin_amdgcn_tensor_load_to_lds(g0, g1, gz4, gz4, gz8, 0);
}

// ---------------------------------------------------------------------------
// K1: relative score term.  S[bh, q, k] = sum_d Q[bh,q,d] * Rk[q,k,d]
// For fixed q this is a GEMM with M = bh; grid = (k64-block, q); 4 waves = 4 bh-tiles.
__global__ void k_rel_scores(const float* __restrict__ q,
                             const float* __restrict__ rk,
                             float* __restrict__ S) {
  const int k0 = blockIdx.x * 64;
  const int qi = blockIdx.y;
  if (k0 > qi) return;  // causal: whole block above diagonal
  const int bh0 = (threadIdx.x >> 5) * 16;

  const float* aq = q + bh0 * QD + qi * ND;   // A[m=bh][d], row stride = Q*D
  v16h a0 = loadA(aq, QD);                    // d = 0..31
  v16h a1 = loadA(aq + 32, QD);               // d = 32..63

  const float* rbase = rk + (size_t)qi * KD + (size_t)k0 * ND;
  __builtin_prefetch(rbase + 64 * ND, 0, 1);  // next k64 block of bigr_k[q]

#pragma unroll
  for (int t = 0; t < 4; ++t) {
    const float* rb = rbase + t * 16 * ND;
    v16h b0 = loadB_kcontig(rb, ND);          // B[d][k], d contiguous
    v16h b1 = loadB_kcontig(rb + 32, ND);
    v8f acc = {};
    acc = WMMA(a0, b0, acc);
    acc = WMMA(a1, b1, acc);
    storeD(S + bh0 * QK + qi * NK + k0 + t * 16, QK, acc);  // m-stride = Q*K (bh)
  }
}

// ---------------------------------------------------------------------------
// K2: content score term, added in place.  S[bh, q, k] += sum_d Q[bh,q,d]*K[bh,k,d]
// M = q; grid = (qt, bh); 4 waves stride over causal k-tiles.  K lives in L2.
__global__ void k_cont_scores(const float* __restrict__ q,
                              const float* __restrict__ kk,
                              float* __restrict__ S) {
  const int qt = blockIdx.x;
  const int bh = blockIdx.y;
  const int w = threadIdx.x >> 5;

  const float* aq = q + bh * QD + qt * 16 * ND;
  v16h a0 = loadA(aq, ND);
  v16h a1 = loadA(aq + 32, ND);

  for (int kt = w; kt <= qt; kt += 4) {       // causal: only kt*16 <= q rows
    const float* kb = kk + bh * KD + kt * 16 * ND;
    v16h b0 = loadB_kcontig(kb, ND);          // B[d][k] = K[k,d]
    v16h b1 = loadB_kcontig(kb + 32, ND);
    v8f acc = {};
    acc = WMMA(a0, b0, acc);
    acc = WMMA(a1, b1, acc);
    storeD_add(S + bh * QK + qt * 16 * NK + kt * 16, NK, acc);
  }
}

// ---------------------------------------------------------------------------
// K3: in-place causal softmax over each (bh,q) row; one wave per row.
__global__ void k_softmax(float* __restrict__ S) {
  const int row = blockIdx.x * 8 + (threadIdx.x >> 5);
  const int lane = threadIdx.x & 31;
  const int qi = row & (NQ - 1);
  float* rp = S + (size_t)row * NK;
  const int kmax = qi + 1;

  float v[32];
  float m = -3.0e38f;
#pragma unroll
  for (int j = 0; j < 32; ++j) {
    const int k = lane + j * 32;
    const float s = (k < kmax) ? rp[k] * 0.125f : -3.0e38f;  // 1/sqrt(64)
    v[j] = s;
    m = fmaxf(m, s);
  }
#pragma unroll
  for (int off = 16; off > 0; off >>= 1) m = fmaxf(m, __shfl_xor(m, off, 32));

  float sum = 0.f;
#pragma unroll
  for (int j = 0; j < 32; ++j) {
    const int k = lane + j * 32;
    const float e = (k < kmax) ? __expf(v[j] - m) : 0.f;
    v[j] = e;
    sum += e;
  }
#pragma unroll
  for (int off = 16; off > 0; off >>= 1) sum += __shfl_xor(sum, off, 32);

  const float inv = 1.0f / sum;
#pragma unroll
  for (int j = 0; j < 32; ++j) rp[lane + j * 32] = v[j] * inv;
}

// ---------------------------------------------------------------------------
// K4a: content output.  out[bh,q,d] = sum_k P[bh,q,k] * V[bh,k,d]
// M = q; grid = (qt, bh); 4 waves = 4 d-tiles.
// V tiles staged via TDM into double-buffered LDS, shared by all 4 waves.
__global__ void k_out_content(const float* __restrict__ P,
                              const float* __restrict__ V,
                              float* __restrict__ out) {
  __shared__ float tile[2][32 * 64];
  const int qt = blockIdx.x;
  const int bh = blockIdx.y;
  const int w = threadIdx.x >> 5;
  const int d0 = w * 16;
  const int kend = (qt + 1) * 16;             // p == 0 beyond the diagonal
  const int nsteps = (kend + 31) / 32;
  const float* vbase = V + bh * KD;

  if (w == 0)
    tdm_load_32x64_f32((unsigned)(size_t)&tile[0][0], vbase, 1024);

  v8f acc = {};
  for (int i = 0; i < nsteps; ++i) {
    const int k0 = i * 32;
    if (w == 0) {
      if (i + 1 < nsteps) {
        tdm_load_32x64_f32((unsigned)(size_t)&tile[(i + 1) & 1][0],
                           vbase + (k0 + 32) * ND, 1024 - (k0 + 32));
        __builtin_amdgcn_s_wait_tensorcnt(1);  // in-order: tile i resident
      } else {
        __builtin_amdgcn_s_wait_tensorcnt(0);
      }
    }
    __syncthreads();                           // tile[i&1] visible to all waves
    v16h a = loadA(P + bh * QK + qt * 16 * NK + k0, NK);  // A[q][k]
    v16h b = loadB_tile(tile[i & 1], d0);                 // B[k][d] from LDS
    acc = WMMA(a, b, acc);
    __syncthreads();                           // done reading before TDM reuses buffer
  }
  storeD(out + bh * QD + qt * 16 * ND + d0, ND, acc);
}

// ---------------------------------------------------------------------------
// K4b: relative output, added in place.  out[bh,q,d] += sum_k P[bh,q,k]*Rv[q,k,d]
// For fixed q this is a GEMM with M = bh; grid = (q); 4 waves = 4 bh-tiles.
// bigr_v[q] (the HBM-streamed operand) staged once per WG via TDM double buffer.
__global__ void k_out_rel(const float* __restrict__ P,
                          const float* __restrict__ rv,
                          float* __restrict__ out) {
  __shared__ float tile[2][32 * 64];
  const int qi = blockIdx.x;
  const int w = threadIdx.x >> 5;
  const int bh0 = w * 16;
  const float* prow = P + bh0 * QK + qi * NK;      // A[m=bh][k], row stride Q*K
  const float* rbase = rv + (size_t)qi * KD;
  const int nsteps = (qi + 1 + 31) / 32;           // p == 0 beyond the diagonal

  if (w == 0)
    tdm_load_32x64_f32((unsigned)(size_t)&tile[0][0], rbase, 1024);

  v8f acc[4] = {{}, {}, {}, {}};
  for (int i = 0; i < nsteps; ++i) {
    const int k0 = i * 32;
    if (w == 0) {
      if (i + 1 < nsteps) {
        tdm_load_32x64_f32((unsigned)(size_t)&tile[(i + 1) & 1][0],
                           rbase + (k0 + 32) * ND, 1024 - (k0 + 32));
        __builtin_amdgcn_s_wait_tensorcnt(1);
      } else {
        __builtin_amdgcn_s_wait_tensorcnt(0);
      }
    }
    __syncthreads();
    v16h a = loadA(prow + k0, QK);
#pragma unroll
    for (int dt = 0; dt < 4; ++dt) {
      v16h b = loadB_tile(tile[i & 1], dt * 16);   // B[k][d] from shared LDS tile
      acc[dt] = WMMA(a, b, acc[dt]);
    }
    __syncthreads();
  }
#pragma unroll
  for (int dt = 0; dt < 4; ++dt)
    storeD_add(out + bh0 * QD + qi * ND + dt * 16, QD, acc[dt]);
}

// ---------------------------------------------------------------------------
extern "C" void kernel_launch(void* const* d_in, const int* in_sizes, int n_in,
                              void* d_out, int out_size, void* d_ws, size_t ws_size,
                              hipStream_t stream) {
  const float* q  = (const float*)d_in[0];  // (4,16,1024,64)
  const float* k  = (const float*)d_in[1];  // (4,16,1024,64)
  const float* v  = (const float*)d_in[2];  // (4,16,1024,64)
  const float* rk = (const float*)d_in[3];  // (1024,1024,64)
  const float* rv = (const float*)d_in[4];  // (1024,1024,64)

  float* out = (float*)d_out;               // output: (4,16,1024,64)
  float* p   = out + BH * QD;               // p_attn: (4,16,1024,1024); doubles as score scratch

  k_rel_scores <<<dim3(16, 1024), 128, 0, stream>>>(q, rk, p);
  k_cont_scores<<<dim3(64, 64),   128, 0, stream>>>(q, k, p);
  k_softmax    <<<dim3(8192),     256, 0, stream>>>(p);
  k_out_content<<<dim3(64, 64),   128, 0, stream>>>(p, v, out);
  k_out_rel    <<<dim3(1024),     128, 0, stream>>>(p, rv, out);
}
